// Qwen3ParoForCausalLM_601295421922
// MI455X (gfx1250) — matrix operand order repaired and checked by reading, not compiled, not run on hardware
//
#include <hip/hip_runtime.h>

// ---------------------------------------------------------------------------
// Qwen3-style layer for MI455X (gfx1250, wave32, WMMA bf16 16x16x32)
// ---------------------------------------------------------------------------

typedef __bf16 bf16_t;
typedef __bf16 v16bf __attribute__((ext_vector_type(16)));
typedef __bf16 v8bf  __attribute__((ext_vector_type(8)));
typedef float  v8f   __attribute__((ext_vector_type(8)));

union U16bf { v16bf v; v8bf h[2]; };

constexpr int   kT    = 2048;
constexpr int   kH    = 2048;
constexpr int   kNH   = 16;
constexpr int   kNKV  = 8;
constexpr int   kHD   = 128;
constexpr int   kI    = 6144;
constexpr float kEPS  = 1e-6f;
constexpr float kSCALE = 0.08838834764831845f;   // 128^-0.5
constexpr float kLN_ROPE_BASE = 13.815510557964274f; // ln(1e6)

// ---------------------------------------------------------------------------
// RMSNorm over a row of n f32 values (block = 256 threads, one row per block)
// ---------------------------------------------------------------------------
__global__ __launch_bounds__(256)
void rmsnorm_kernel(const float* __restrict__ X, const float* __restrict__ w,
                    float* __restrict__ Y, int n) {
  const int t = blockIdx.x;
  const float* x = X + (size_t)t * n;
  float ss = 0.f;
  for (int i = threadIdx.x; i < n; i += 256) { float v = x[i]; ss += v * v; }
  __shared__ float red[8];
  for (int d = 1; d < 32; d <<= 1) ss += __shfl_xor(ss, d, 32);
  if ((threadIdx.x & 31) == 0) red[threadIdx.x >> 5] = ss;
  __syncthreads();
  float tot = 0.f;
#pragma unroll
  for (int wv = 0; wv < 8; ++wv) tot += red[wv];
  const float r = rsqrtf(tot / (float)n + kEPS);
  float* y = Y + (size_t)t * n;
  for (int i = threadIdx.x; i < n; i += 256) y[i] = x[i] * r * w[i];
}

// ---------------------------------------------------------------------------
// resid = A + B ; Y = rmsnorm(resid) * w        (n == 2048, block = 256)
// ---------------------------------------------------------------------------
__global__ __launch_bounds__(256)
void add_rmsnorm_kernel(const float* __restrict__ A, const float* __restrict__ B,
                        const float* __restrict__ w, float* __restrict__ R,
                        float* __restrict__ Y, int n) {
  const int t = blockIdx.x;
  float vals[8];
  float ss = 0.f;
#pragma unroll
  for (int k = 0; k < 8; ++k) {
    int i = k * 256 + threadIdx.x;
    float v = A[(size_t)t * n + i] + B[(size_t)t * n + i];
    vals[k] = v;
    R[(size_t)t * n + i] = v;
    ss += v * v;
  }
  __shared__ float red[8];
  for (int d = 1; d < 32; d <<= 1) ss += __shfl_xor(ss, d, 32);
  if ((threadIdx.x & 31) == 0) red[threadIdx.x >> 5] = ss;
  __syncthreads();
  float tot = 0.f;
#pragma unroll
  for (int wv = 0; wv < 8; ++wv) tot += red[wv];
  const float r = rsqrtf(tot / (float)n + kEPS);
#pragma unroll
  for (int k = 0; k < 8; ++k) {
    int i = k * 256 + threadIdx.x;
    Y[(size_t)t * n + i] = vals[k] * r * w[i];
  }
}

// ---------------------------------------------------------------------------
// Activation transform of rot_linear: y = rotate(x * channel_scales) -> bf16
// One thread per (row t, pair p).
// ---------------------------------------------------------------------------
__global__ __launch_bounds__(256)
void act_rot_kernel(const float* __restrict__ X, const float* __restrict__ cs,
                    const float* __restrict__ theta, const int* __restrict__ pairs,
                    bf16_t* __restrict__ Y, int in_f) {
  const int p = blockIdx.x * 256 + threadIdx.x;
  const int np = in_f >> 1;
  if (p >= np) return;
  const int t = blockIdx.y;
  const int i = pairs[2 * p + 0];
  const int j = pairs[2 * p + 1];
  const float* x = X + (size_t)t * in_f;
  const float xi = x[i] * cs[i];
  const float xj = x[j] * cs[j];
  const float th = theta[p];
  const float c = __cosf(th), s = __sinf(th);
  bf16_t* y = Y + (size_t)t * in_f;
  y[i] = (bf16_t)(c * xi - s * xj);
  y[j] = (bf16_t)(s * xi + c * xj);
}

// ---------------------------------------------------------------------------
// Dequant-fused blocked GEMM: C[T,out_f] = A_bf16[T,in_f] * W^T
// W[n][k] = (qweight[n][k] - qzeros[n][k/128]) * scales[n][k/128]
//
// Block = 256 threads (8 waves) computes a 128x128 C tile; per K-step (32):
//   - cooperative load A 128x32 bf16 tile into LDS
//   - cooperative dequant W 128x32 (int32 -> bf16) into LDS
//   - each wave: 64(M) x 32(N) sub-tile = 8 x v_wmma_f32_16x16x32_bf16
// LDS rows padded to 40 elements to break bank periodicity.
// Grid: (out_f/128, T/128).
// ---------------------------------------------------------------------------
constexpr int kLdsPitch = 40;   // bf16 elements per LDS row (32 data + 8 pad)

__global__ __launch_bounds__(256)
void gemm_dq_kernel(const bf16_t* __restrict__ A, const int* __restrict__ qw,
                    const int* __restrict__ qz, const float* __restrict__ sc,
                    float* __restrict__ C, int in_f, int out_f) {
  __shared__ __align__(16) bf16_t As[128 * kLdsPitch];
  __shared__ __align__(16) bf16_t Bs[128 * kLdsPitch];

  const int tid  = threadIdx.x;
  const int lane = tid & 31;
  const int wave = tid >> 5;
  const int half = lane >> 4;
  const int l16  = lane & 15;

  const int m0 = blockIdx.y * 128;
  const int n0 = blockIdx.x * 128;
  const int groups = in_f >> 7;            // in_f / 128

  // cooperative-load assignment: 2 threads per 32-wide row (16 elements each)
  const int ldrow = tid >> 1;              // 0..127
  const int ldk   = (tid & 1) * 16;        // 0 or 16

  const bf16_t* agp  = A  + (size_t)(m0 + ldrow) * in_f + ldk;
  const int*    wgp  = qw + (size_t)(n0 + ldrow) * in_f + ldk;
  const int*    zrow = qz + (size_t)(n0 + ldrow) * groups;
  const float*  srow = sc + (size_t)(n0 + ldrow) * groups;

  // wave sub-tile: 64 rows (M) x 32 cols (N)
  const int mw = (wave >> 2) * 64;
  const int nw = (wave & 3) * 32;

  v8f acc[4][2];
#pragma unroll
  for (int mt = 0; mt < 4; ++mt)
#pragma unroll
    for (int ns = 0; ns < 2; ++ns)
#pragma unroll
      for (int r = 0; r < 8; ++r) acc[mt][ns][r] = 0.f;

  for (int k0 = 0; k0 < in_f; k0 += 32) {
    __syncthreads();                       // previous compute done before overwrite

    // ---- stage A tile (16 bf16 = 32B per thread)
    {
      const int4* g = (const int4*)(agp + k0);
      int4* d = (int4*)(As + ldrow * kLdsPitch + ldk);
      d[0] = g[0];
      d[1] = g[1];
    }
    // ---- stage dequantized B tile (16 weights per thread)
    {
      const int gi = k0 >> 7;              // group index (constant across 32-K step)
      const float z = (float)zrow[gi];
      const float s = srow[gi];
      int qv[16];
      *(int4*)(qv + 0)  = *(const int4*)(wgp + k0 + 0);
      *(int4*)(qv + 4)  = *(const int4*)(wgp + k0 + 4);
      *(int4*)(qv + 8)  = *(const int4*)(wgp + k0 + 8);
      *(int4*)(qv + 12) = *(const int4*)(wgp + k0 + 12);
      bf16_t* d = Bs + ldrow * kLdsPitch + ldk;
#pragma unroll
      for (int e = 0; e < 16; ++e) d[e] = (bf16_t)(((float)qv[e] - z) * s);
    }
    // ---- prefetch next weight K-tile (global_prefetch_b8, speculative-safe)
    if (k0 + 32 < in_f) __builtin_prefetch(wgp + k0 + 32, 0, 3);

    __syncthreads();                       // tiles visible to all waves

    // ---- build fragments from LDS
    U16bf a[4], b[2];
#pragma unroll
    for (int mt = 0; mt < 4; ++mt) {
      const bf16_t* p = As + (mw + mt * 16 + l16) * kLdsPitch;
      a[mt].h[0] = *(const v8bf*)(p + half * 8);
      a[mt].h[1] = *(const v8bf*)(p + 16 + half * 8);
    }
#pragma unroll
    for (int ns = 0; ns < 2; ++ns) {
      const bf16_t* p = Bs + (nw + ns * 16 + l16) * kLdsPitch + half * 8;
      b[ns].h[0] = *(const v8bf*)(p);
      b[ns].h[1] = *(const v8bf*)(p + 16);
    }

    // ---- 8 WMMAs: 64x32 sub-tile
#pragma unroll
    for (int mt = 0; mt < 4; ++mt)
#pragma unroll
      for (int ns = 0; ns < 2; ++ns)
        acc[mt][ns] = __builtin_amdgcn_wmma_f32_16x16x32_bf16(
            false, a[mt].v, false, b[ns].v, (short)0, acc[mt][ns], false, false);
  }

  // ---- D layout per 16x16 tile: M = r + 8*half, N = l16
#pragma unroll
  for (int mt = 0; mt < 4; ++mt)
#pragma unroll
    for (int ns = 0; ns < 2; ++ns) {
      float* cp = C + (size_t)(m0 + mw + mt * 16 + half * 8) * out_f
                    + n0 + nw + ns * 16 + l16;
#pragma unroll
      for (int r = 0; r < 8; ++r) cp[(size_t)r * out_f] = acc[mt][ns][r];
    }
}

// ---------------------------------------------------------------------------
// Per-head RMSNorm + RoPE, f32 in -> bf16 out.  Block = 128 (one (t,head)).
// ---------------------------------------------------------------------------
__global__ __launch_bounds__(128)
void norm_rope_kernel(const float* __restrict__ X, const float* __restrict__ w,
                      const int* __restrict__ pos, bf16_t* __restrict__ Y,
                      int nheads) {
  const int t = blockIdx.x / nheads;
  const int h = blockIdx.x % nheads;
  const int d = threadIdx.x;
  __shared__ float xs[128];
  __shared__ float red[4];
  const float* xp = X + ((size_t)t * nheads + h) * kHD;
  float x = xp[d];
  float ss = x * x;
  for (int m = 1; m < 32; m <<= 1) ss += __shfl_xor(ss, m, 32);
  if ((threadIdx.x & 31) == 0) red[threadIdx.x >> 5] = ss;
  __syncthreads();
  const float tot = red[0] + red[1] + red[2] + red[3];
  const float r = rsqrtf(tot / (float)kHD + kEPS);
  xs[d] = x * r * w[d];
  __syncthreads();
  if (d < 64) {
    const float p   = (float)pos[t];
    const float inv = __expf(-(float)d * (kLN_ROPE_BASE / 64.f));
    const float ang = p * inv;
    const float c = __cosf(ang), s = __sinf(ang);
    const float x1 = xs[d], x2 = xs[d + 64];
    bf16_t* yp = Y + ((size_t)t * nheads + h) * kHD;
    yp[d]      = (bf16_t)(x1 * c - x2 * s);
    yp[d + 64] = (bf16_t)(x1 * s + x2 * c);
  }
}

// ---------------------------------------------------------------------------
// V f32 [T, NKV*HD] -> VT bf16 [NKV*HD, T]  (transposed for P*V B-fragments)
// ---------------------------------------------------------------------------
__global__ __launch_bounds__(256)
void vt_kernel(const float* __restrict__ V, bf16_t* __restrict__ VT) {
  const int idx = blockIdx.x * 256 + threadIdx.x;   // over T * 1024
  const int t = idx >> 10;
  const int n = idx & 1023;
  VT[(size_t)n * kT + t] = (bf16_t)V[idx];
}

// ---------------------------------------------------------------------------
// Causal flash attention with WMMA bf16.
// Wave per (head, 16-query block); key blocks of 32.
// Q bf16 [T,NH,HD], K bf16 [T,NKV,HD], VT bf16 [NKV*HD, T], O f32 [T, NH*HD]
// ---------------------------------------------------------------------------
__global__ __launch_bounds__(256)
void attn_kernel(const bf16_t* __restrict__ Q, const bf16_t* __restrict__ K,
                 const bf16_t* __restrict__ VT, float* __restrict__ O) {
  __shared__ __align__(16) bf16_t pshare[8][16 * 32];
  const int lane = threadIdx.x & 31;
  const int wave = threadIdx.x >> 5;
  const int half = lane >> 4;
  const int l16  = lane & 15;
  const int wid  = blockIdx.x * 8 + wave;
  const int head = wid >> 7;            // / (T/16)
  const int q0   = (wid & 127) << 4;
  const int kv   = head >> 1;           // NH/NKV == 2

  // Q fragments for 4 K-steps of 32 along HD
  U16bf aq[4];
  const bf16_t* qrow = Q + ((size_t)(q0 + l16) * kNH + head) * kHD;
#pragma unroll
  for (int ks = 0; ks < 4; ++ks) {
    aq[ks].h[0] = *(const v8bf*)(qrow + ks * 32 + half * 8);
    aq[ks].h[1] = *(const v8bf*)(qrow + ks * 32 + 16 + half * 8);
  }

  v8f o[8];
#pragma unroll
  for (int nt = 0; nt < 8; ++nt)
#pragma unroll
    for (int r = 0; r < 8; ++r) o[nt][r] = 0.f;
  float rowM[8], rowL[8];
#pragma unroll
  for (int r = 0; r < 8; ++r) { rowM[r] = -3.0e38f; rowL[r] = 0.f; }

  bf16_t* pbuf = pshare[wave];

  for (int s0 = 0; s0 <= q0 + 15; s0 += 32) {
    // ---- S = Q * K^T  (16 queries x 32 keys) as two 16x16 tiles
    v8f c0, c1;
#pragma unroll
    for (int r = 0; r < 8; ++r) { c0[r] = 0.f; c1[r] = 0.f; }
#pragma unroll
    for (int ks = 0; ks < 4; ++ks) {
      U16bf bk0, bk1;
      const bf16_t* kr0 = K + ((size_t)(s0 + l16) * kNKV + kv) * kHD + ks * 32 + half * 8;
      const bf16_t* kr1 = K + ((size_t)(s0 + 16 + l16) * kNKV + kv) * kHD + ks * 32 + half * 8;
      bk0.h[0] = *(const v8bf*)(kr0);
      bk0.h[1] = *(const v8bf*)(kr0 + 16);
      bk1.h[0] = *(const v8bf*)(kr1);
      bk1.h[1] = *(const v8bf*)(kr1 + 16);
      c0 = __builtin_amdgcn_wmma_f32_16x16x32_bf16(false, aq[ks].v, false, bk0.v,
                                                   (short)0, c0, false, false);
      c1 = __builtin_amdgcn_wmma_f32_16x16x32_bf16(false, aq[ks].v, false, bk1.v,
                                                   (short)0, c1, false, false);
    }

    // ---- mask + online softmax (half-waves own disjoint row sets: M = 8*half + r)
#pragma unroll
    for (int r = 0; r < 8; ++r) {
      const int qi = q0 + half * 8 + r;
      float v0 = c0[r] * kSCALE;
      float v1 = c1[r] * kSCALE;
      if (s0 + l16 > qi)      v0 = -1.0e30f;
      if (s0 + 16 + l16 > qi) v1 = -1.0e30f;
      float mx = fmaxf(v0, v1);
      for (int m = 1; m < 16; m <<= 1) mx = fmaxf(mx, __shfl_xor(mx, m, 16));
      const float Mn  = fmaxf(rowM[r], mx);
      const float fac = __expf(rowM[r] - Mn);
      rowM[r] = Mn;
      const float p0 = __expf(v0 - Mn);
      const float p1 = __expf(v1 - Mn);
      float sm = p0 + p1;
      for (int m = 1; m < 16; m <<= 1) sm += __shfl_xor(sm, m, 16);
      rowL[r] = rowL[r] * fac + sm;
#pragma unroll
      for (int nt = 0; nt < 8; ++nt) o[nt][r] *= fac;
      const int row = half * 8 + r;
      pbuf[row * 32 + l16]      = (bf16_t)p0;
      pbuf[row * 32 + 16 + l16] = (bf16_t)p1;
    }
    __builtin_amdgcn_wave_barrier();   // same-wave LDS: DS ops are in order

    // ---- P fragment (A layout, 16x32)
    U16bf ap;
    const bf16_t* prow = pbuf + l16 * 32;
    ap.h[0] = *(const v8bf*)(prow + half * 8);
    ap.h[1] = *(const v8bf*)(prow + 16 + half * 8);

    // ---- O += P * V  (8 hd-tiles of 16)
#pragma unroll
    for (int nt = 0; nt < 8; ++nt) {
      U16bf bv;
      const bf16_t* vr = VT + ((size_t)kv * kHD + nt * 16 + l16) * kT + s0 + half * 8;
      bv.h[0] = *(const v8bf*)(vr);
      bv.h[1] = *(const v8bf*)(vr + 16);
      o[nt] = __builtin_amdgcn_wmma_f32_16x16x32_bf16(false, ap.v, false, bv.v,
                                                      (short)0, o[nt], false, false);
    }
    __builtin_amdgcn_wave_barrier();
  }

  // ---- normalize and store O (f32, [T, NH*HD])
#pragma unroll
  for (int nt = 0; nt < 8; ++nt) {
#pragma unroll
    for (int r = 0; r < 8; ++r) {
      const int row = q0 + half * 8 + r;
      O[(size_t)row * (kNH * kHD) + head * kHD + nt * 16 + l16] = o[nt][r] / rowL[r];
    }
  }
}

// ---------------------------------------------------------------------------
// SwiGLU: G <- silu(G) * U    (in place on G)
// ---------------------------------------------------------------------------
__global__ __launch_bounds__(256)
void silu_mul_kernel(float* __restrict__ G, const float* __restrict__ U) {
  const size_t i = (size_t)blockIdx.x * 256 + threadIdx.x;
  const float g = G[i];
  G[i] = g * (1.0f / (1.0f + __expf(-g))) * U[i];
}

__global__ __launch_bounds__(256)
void add_kernel(const float* __restrict__ A, const float* __restrict__ B,
                float* __restrict__ Y) {
  const size_t i = (size_t)blockIdx.x * 256 + threadIdx.x;
  Y[i] = A[i] + B[i];
}

// ---------------------------------------------------------------------------
// Host side
// ---------------------------------------------------------------------------
enum {
  IN_HS = 0, IN_POS = 1,
  FP_Q_THETA = 2,  FP_Q_CS,  FP_Q_SC,
  FP_K_THETA,      FP_K_CS,  FP_K_SC,
  FP_V_THETA,      FP_V_CS,  FP_V_SC,
  FP_O_THETA,      FP_O_CS,  FP_O_SC,
  FP_G_THETA,      FP_G_CS,  FP_G_SC,
  FP_U_THETA,      FP_U_CS,  FP_U_SC,
  FP_D_THETA,      FP_D_CS,  FP_D_SC,
  FP_INLN, FP_POSTLN, FP_QNORM, FP_KNORM,
  IP_Q_QW, IP_Q_QZ, IP_Q_PR,
  IP_K_QW, IP_K_QZ, IP_K_PR,
  IP_V_QW, IP_V_QZ, IP_V_PR,
  IP_O_QW, IP_O_QZ, IP_O_PR,
  IP_G_QW, IP_G_QZ, IP_G_PR,
  IP_U_QW, IP_U_QZ, IP_U_PR,
  IP_D_QW, IP_D_QZ, IP_D_PR,
};

extern "C" void kernel_launch(void* const* d_in, const int* in_sizes, int n_in,
                              void* d_out, int out_size, void* d_ws, size_t ws_size,
                              hipStream_t stream) {
  (void)in_sizes; (void)n_in; (void)out_size; (void)ws_size;
  auto Fp = [&](int i) { return (const float*)d_in[i]; };
  auto Ip = [&](int i) { return (const int*)d_in[i]; };

  char* ws = (char*)d_ws;
  size_t off = 0;
  auto carve = [&](size_t bytes) {
    void* p = (void*)(ws + off);
    off += (bytes + 255) & ~(size_t)255;
    return p;
  };

  float*  hsln  = (float*) carve((size_t)kT * kH * 4);          // normed input
  bf16_t* abuf  = (bf16_t*)carve((size_t)kT * kI * 2);          // shared bf16 activation
  float*  qf    = (float*) carve((size_t)kT * kNH  * kHD * 4);  // q / o / down (reused)
  float*  kf    = (float*) carve((size_t)kT * kNKV * kHD * 4);
  float*  vf    = (float*) carve((size_t)kT * kNKV * kHD * 4);
  bf16_t* qb    = (bf16_t*)carve((size_t)kT * kNH  * kHD * 2);
  bf16_t* kb    = (bf16_t*)carve((size_t)kT * kNKV * kHD * 2);
  bf16_t* vtb   = (bf16_t*)carve((size_t)kT * kNKV * kHD * 2);
  float*  attf  = (float*) carve((size_t)kT * kNH  * kHD * 4);
  float*  r2    = (float*) carve((size_t)kT * kH * 4);
  float*  hs2   = (float*) carve((size_t)kT * kH * 4);
  float*  gatef = (float*) carve((size_t)kT * kI * 4);
  float*  upf   = (float*) carve((size_t)kT * kI * 4);

  auto linear = [&](const float* xin, int fp0, int ip0, int in_f, int out_f,
                    float* cout) {
    dim3 g1((in_f / 2 + 255) / 256, kT);
    act_rot_kernel<<<g1, 256, 0, stream>>>(xin, Fp(fp0 + 1), Fp(fp0 + 0),
                                           Ip(ip0 + 2), abuf, in_f);
    dim3 g2(out_f / 128, kT / 128);
    gemm_dq_kernel<<<g2, 256, 0, stream>>>(abuf, Ip(ip0 + 0), Ip(ip0 + 1),
                                           Fp(fp0 + 2), cout, in_f, out_f);
  };

  // 1) input RMSNorm
  rmsnorm_kernel<<<kT, 256, 0, stream>>>(Fp(IN_HS), Fp(FP_INLN), hsln, kH);

  // 2) q / k / v projections (dequant-fused WMMA GEMMs)
  linear(hsln, FP_Q_THETA, IP_Q_QW, kH, kNH * kHD, qf);
  linear(hsln, FP_K_THETA, IP_K_QW, kH, kNKV * kHD, kf);
  linear(hsln, FP_V_THETA, IP_V_QW, kH, kNKV * kHD, vf);

  // 3) head RMSNorm + RoPE -> bf16; V transposed copy for attention
  norm_rope_kernel<<<kT * kNH, 128, 0, stream>>>(qf, Fp(FP_QNORM), Ip(IN_POS), qb, kNH);
  norm_rope_kernel<<<kT * kNKV, 128, 0, stream>>>(kf, Fp(FP_KNORM), Ip(IN_POS), kb, kNKV);
  vt_kernel<<<(kT * kNKV * kHD) / 256, 256, 0, stream>>>(vf, vtb);

  // 4) causal flash attention (WMMA)
  attn_kernel<<<(kNH * (kT / 16)) / 8, 256, 0, stream>>>(qb, kb, vtb, attf);

  // 5) output projection + residual + post-LN
  linear(attf, FP_O_THETA, IP_O_QW, kNH * kHD, kH, qf);
  add_rmsnorm_kernel<<<kT, 256, 0, stream>>>(qf, Fp(IN_HS), Fp(FP_POSTLN), r2, hs2, kH);

  // 6) MLP: gate / up, SwiGLU, down, final residual
  linear(hs2, FP_G_THETA, IP_G_QW, kH, kI, gatef);
  linear(hs2, FP_U_THETA, IP_U_QW, kH, kI, upf);
  silu_mul_kernel<<<(kT * (kI / 256)), 256, 0, stream>>>(gatef, upf);
  linear(gatef, FP_D_THETA, IP_D_QW, kI, kH, qf);
  add_kernel<<<(kT * kH) / 256, 256, 0, stream>>>(qf, r2, (float*)d_out);
}
